// BasicBlock_49830210568448
// MI455X (gfx1250) — 8-bit matrix operand order repaired and checked by reading, compiled, not run on hardware
//
#include <hip/hip_runtime.h>

typedef __attribute__((ext_vector_type(8))) int v8i;

namespace {
constexpr int kC  = 128;
constexpr int kH  = 56;
constexpr int kW  = 56;
constexpr int kHW = kH * kW;
constexpr int kN  = 64;
constexpr int kM  = kN * kHW;
constexpr int kKsteps = (kC * 9) / 64;
constexpr int kWElems = kC * kC * 9;
constexpr int kElems  = kM * kC;
constexpr int kMB     = 128;
}

__global__ void zero_u32_kernel(unsigned int* __restrict__ p, int n) {
    int i = blockIdx.x * blockDim.x + threadIdx.x;
    if (i < n) p[i] = 0u;
}

__global__ void wmax_kernel(const float* __restrict__ w, int n,
                            unsigned int* __restrict__ amax_bits) {
    __shared__ float red[256];
    int i = blockIdx.x * blockDim.x + threadIdx.x;
    float m = 0.0f;
    if (i < n) m = fabsf(w[i]);
    red[threadIdx.x] = m;
    __syncthreads();
    for (int s = 128; s > 0; s >>= 1) {
        if (threadIdx.x < s)
            red[threadIdx.x] = fmaxf(red[threadIdx.x], red[threadIdx.x + s]);
        __syncthreads();
    }
    if (threadIdx.x == 0) atomicMax(amax_bits, __float_as_uint(red[0]));
}

__global__ void prepack_kernel(const float* __restrict__ w,
                               const unsigned int* __restrict__ amax_bits,
                               signed char* __restrict__ wq) {
    int i = blockIdx.x * blockDim.x + threadIdx.x;
    if (i >= kWElems) return;
    int o  = i / (9 * kC);
    int k  = i % (9 * kC);
    int rs = k / kC;
    int c  = k % kC;
    int r  = rs / 3, s = rs % 3;
    float alpha = __uint_as_float(*amax_bits) + 1e-12f;
    float v = w[((size_t)(o * kC + c) * 3 + r) * 3 + s] / alpha;
    v = fminf(fmaxf(v, -1.0f), 1.0f);
    wq[i] = (signed char)__float2int_rn(v * 7.0f);
}

__global__ void actquant_x_kernel(const float* __restrict__ x,
                                  unsigned char* __restrict__ xq) {
    size_t t = (size_t)blockIdx.x * blockDim.x + threadIdx.x;
    size_t i = t * 4;
    if (i >= (size_t)kElems) return;
    const int hw = (int)(i % kHW);
    const int c  = (int)((i / kHW) % kC);
    const int n  = (int)(i / ((size_t)kHW * kC));
    const float4 v = *(const float4*)(x + i);
    unsigned char* q = xq + ((size_t)n * kHW + hw) * kC + c;
    q[0 * kC] = (unsigned char)__float2int_rn(fminf(fmaxf(v.x, 0.0f), 1.0f) * 15.0f);
    q[1 * kC] = (unsigned char)__float2int_rn(fminf(fmaxf(v.y, 0.0f), 1.0f) * 15.0f);
    q[2 * kC] = (unsigned char)__float2int_rn(fminf(fmaxf(v.z, 0.0f), 1.0f) * 15.0f);
    q[3 * kC] = (unsigned char)__float2int_rn(fminf(fmaxf(v.w, 0.0f), 1.0f) * 15.0f);
}

__device__ __forceinline__ void stage_fetch2(const unsigned char* __restrict__ xq,
                                             int an, int ah, int aw, int aoff,
                                             int kk, uint4& lo, uint4& hi) {
    const int rs = kk >> 1;
    const int r  = rs / 3, s = rs % 3;
    const int cb = (kk & 1) * 64;
    const int hh = ah + r - 1;
    const int ww = aw + s - 1;
    lo = uint4{0u, 0u, 0u, 0u};
    hi = uint4{0u, 0u, 0u, 0u};
    if (hh >= 0 && hh < kH && ww >= 0 && ww < kW) {
        const unsigned char* p =
            xq + ((size_t)(an * kH + hh) * kW + ww) * kC + cb + aoff;
        lo = *(const uint4*)(p);
        hi = *(const uint4*)(p + 16);
    }
}

__global__ __launch_bounds__(256)
void conv3x3_q_wmma_kernel(const unsigned char* __restrict__ xq,
                           const signed char* __restrict__ wq,
                           const unsigned int* __restrict__ amax_bits,
                           float* __restrict__ y,
                           float* __restrict__ gsum,
                           float* __restrict__ gsumsq) {
    __shared__ unsigned char As[2][kMB * 64];
    __shared__ float lsum[kC];
    __shared__ float lsumsq[kC];

    const int tid  = threadIdx.x;
    const int lane = tid & 31;
    const int wave = tid >> 5;
    const int m0   = blockIdx.x * kMB;

    if (tid < kC) { lsum[tid] = 0.0f; lsumsq[tid] = 0.0f; }

    const int arow = tid >> 1;
    const int aoff = (tid & 1) * 32;
    const int am   = m0 + arow;
    const int an   = am / kHW;
    const int ahw  = am % kHW;
    const int ah   = ahw / kW;
    const int aw   = ahw % kW;

    const int mrow   = lane & 15;
    const int ka_off = (lane & 16) ? 8 : 0;
    const int kb_off = (lane & 16) ? 8 : 0;
    const int och    = wave * 16 + (lane & 15);
    const signed char* brow = wq + (size_t)och * (9 * kC);

    v8i acc[8] = {};

    {
        uint4 p0, p1;
        stage_fetch2(xq, an, ah, aw, aoff, 0, p0, p1);
        *(uint4*)(As[0] + arow * 64 + aoff)      = p0;
        *(uint4*)(As[0] + arow * 64 + aoff + 16) = p1;
    }
    __syncthreads();

#pragma unroll
    for (int kk = 0; kk < kKsteps; ++kk) {
        uint4 n0 = {0u, 0u, 0u, 0u}, n1 = {0u, 0u, 0u, 0u};
        if (kk + 1 < kKsteps) stage_fetch2(xq, an, ah, aw, aoff, kk + 1, n0, n1);

        const unsigned char* buf = As[kk & 1];

        v8i bfrag;
        const signed char* bp = brow + kk * 64;
#pragma unroll
        for (int v = 0; v < 8; ++v) {
            const int kbase = (v >> 1) * 16 + (v & 1) * 4 + kb_off;
            bfrag[v] = *(const int*)(bp + kbase);
        }

#pragma unroll
        for (int mi = 0; mi < 8; ++mi) {
            v8i afrag;
            const unsigned char* ap = buf + (mi * 16 + mrow) * 64;
#pragma unroll
            for (int v = 0; v < 8; ++v) {
                const int kbase = (v >> 1) * 16 + (v & 1) * 4 + ka_off;
                afrag[v] = *(const int*)(ap + kbase);
            }
            acc[mi] = __builtin_amdgcn_wmma_i32_16x16x64_iu8(
                 false, afrag,  true, bfrag, acc[mi],
                 false,  false);
        }

        if (kk + 1 < kKsteps) {
            unsigned char* nbuf = As[(kk + 1) & 1];
            *(uint4*)(nbuf + arow * 64 + aoff)      = n0;
            *(uint4*)(nbuf + arow * 64 + aoff + 16) = n1;
        }
        __syncthreads();
    }

    const float scale = (__uint_as_float(*amax_bits) + 1e-12f) * (1.0f / 105.0f);
    const int mbase = (lane & 16) ? 8 : 0;
    float psum = 0.0f, psumsq = 0.0f;
#pragma unroll
    for (int mi = 0; mi < 8; ++mi) {
#pragma unroll
        for (int d = 0; d < 8; ++d) {
            const int M = mi * 16 + d + mbase;
            const float val = (float)acc[mi][d] * scale;
            y[(size_t)(m0 + M) * kC + och] = val;
            psum += val;
            psumsq += val * val;
        }
    }
    atomicAdd(&lsum[och], psum);
    atomicAdd(&lsumsq[och], psumsq);
    __syncthreads();
    if (tid < kC) {
        atomicAdd(&gsum[tid], lsum[tid]);
        atomicAdd(&gsumsq[tid], lsumsq[tid]);
    }
}

__global__ void bn_finalize_kernel(const float* __restrict__ gsum,
                                   const float* __restrict__ gsumsq,
                                   float* __restrict__ mean,
                                   float* __restrict__ rstd) {
    int c = threadIdx.x;
    if (c < kC) {
        const float inv = 1.0f / (float)kM;
        float mu  = gsum[c] * inv;
        float var = gsumsq[c] * inv - mu * mu;
        mean[c] = mu;
        rstd[c] = rsqrtf(var + 1e-5f);
    }
}

__global__ void bn_actquant_kernel(const float* __restrict__ y,
                                   const float* __restrict__ mean,
                                   const float* __restrict__ rstd,
                                   const float* __restrict__ gamma,
                                   const float* __restrict__ beta,
                                   unsigned char* __restrict__ xq) {
    size_t t = (size_t)blockIdx.x * blockDim.x + threadIdx.x;
    size_t i = t * 4;
    if (i >= (size_t)kElems) return;
    const int c = (int)(i & (kC - 1));
    const float4 v  = *(const float4*)(y + i);
    const float4 mu = *(const float4*)(mean + c);
    const float4 rs = *(const float4*)(rstd + c);
    const float4 ga = *(const float4*)(gamma + c);
    const float4 be = *(const float4*)(beta + c);
    uchar4 q;
    q.x = (unsigned char)__float2int_rn(fminf(fmaxf((v.x - mu.x) * rs.x * ga.x + be.x, 0.0f), 1.0f) * 15.0f);
    q.y = (unsigned char)__float2int_rn(fminf(fmaxf((v.y - mu.y) * rs.y * ga.y + be.y, 0.0f), 1.0f) * 15.0f);
    q.z = (unsigned char)__float2int_rn(fminf(fmaxf((v.z - mu.z) * rs.z * ga.z + be.z, 0.0f), 1.0f) * 15.0f);
    q.w = (unsigned char)__float2int_rn(fminf(fmaxf((v.w - mu.w) * rs.w * ga.w + be.w, 0.0f), 1.0f) * 15.0f);
    *(uchar4*)(xq + i) = q;
}

__global__ void bn_add_out_kernel(const float* __restrict__ y,
                                  const float* __restrict__ mean,
                                  const float* __restrict__ rstd,
                                  const float* __restrict__ gamma,
                                  const float* __restrict__ beta,
                                  const float* __restrict__ x,
                                  float* __restrict__ out) {
    size_t t = (size_t)blockIdx.x * blockDim.x + threadIdx.x;
    size_t i = t * 4;
    if (i >= (size_t)kElems) return;
    const int    hw = (int)(i % kHW);
    const int    c  = (int)((i / kHW) % kC);
    const size_t n  = i / ((size_t)kHW * kC);
    const float mu = mean[c], rsd = rstd[c], ga = gamma[c], be = beta[c];
    const float scale = rsd * ga;
    const float shift = be - mu * scale;
    const float* yp = y + (n * kHW + hw) * kC + c;
    const float4 xi = *(const float4*)(x + i);
    float4 o;
    o.x = yp[0 * kC] * scale + shift + xi.x;
    o.y = yp[1 * kC] * scale + shift + xi.y;
    o.z = yp[2 * kC] * scale + shift + xi.z;
    o.w = yp[3 * kC] * scale + shift + xi.w;
    *(float4*)(out + i) = o;
}

extern "C" void kernel_launch(void* const* d_in, const int* in_sizes, int n_in,
                              void* d_out, int out_size, void* d_ws, size_t ws_size,
                              hipStream_t stream) {
    const float* x      = (const float*)d_in[0];
    const float* w1     = (const float*)d_in[1];
    const float* w2     = (const float*)d_in[2];
    const float* gamma1 = (const float*)d_in[3];
    const float* beta1  = (const float*)d_in[4];
    const float* gamma2 = (const float*)d_in[5];
    const float* beta2  = (const float*)d_in[6];
    float* out = (float*)d_out;

    char* ws = (char*)d_ws;
    float*        gsum1   = (float*)(ws + 0);
    float*        gsumsq1 = (float*)(ws + 512);
    float*        gsum2   = (float*)(ws + 1024);
    float*        gsumsq2 = (float*)(ws + 1536);
    float*        mean1   = (float*)(ws + 2048);
    float*        rstd1   = (float*)(ws + 2560);
    float*        mean2   = (float*)(ws + 3072);
    float*        rstd2   = (float*)(ws + 3584);
    unsigned int* amax1   = (unsigned int*)(ws + 4096);
    unsigned int* amax2   = (unsigned int*)(ws + 4160);
    signed char*  wq1     = (signed char*)(ws + 4352);
    signed char*  wq2     = wq1 + kWElems;
    unsigned char* xq     = (unsigned char*)(ws + 4352 + 2 * kWElems);
    float*        yraw    = (float*)(ws + 4352 + 2 * kWElems + kElems);

    const int v4_blocks = (kElems / 4 + 255) / 256;
    const int w_blocks  = (kWElems + 255) / 256;
    const int m_blocks  = kM / kMB;

    zero_u32_kernel<<<5, 256, 0, stream>>>((unsigned int*)ws, 4352 / 4);

    wmax_kernel<<<w_blocks, 256, 0, stream>>>(w1, kWElems, amax1);
    wmax_kernel<<<w_blocks, 256, 0, stream>>>(w2, kWElems, amax2);
    prepack_kernel<<<w_blocks, 256, 0, stream>>>(w1, amax1, wq1);
    prepack_kernel<<<w_blocks, 256, 0, stream>>>(w2, amax2, wq2);

    actquant_x_kernel<<<v4_blocks, 256, 0, stream>>>(x, xq);

    conv3x3_q_wmma_kernel<<<m_blocks, 256, 0, stream>>>(xq, wq1, amax1, yraw,
                                                        gsum1, gsumsq1);
    bn_finalize_kernel<<<1, 128, 0, stream>>>(gsum1, gsumsq1, mean1, rstd1);

    bn_actquant_kernel<<<v4_blocks, 256, 0, stream>>>(yraw, mean1, rstd1,
                                                      gamma1, beta1, xq);

    conv3x3_q_wmma_kernel<<<m_blocks, 256, 0, stream>>>(xq, wq2, amax2, yraw,
                                                        gsum2, gsumsq2);
    bn_finalize_kernel<<<1, 128, 0, stream>>>(gsum2, gsumsq2, mean2, rstd2);

    bn_add_out_kernel<<<v4_blocks, 256, 0, stream>>>(yraw, mean2, rstd2,
                                                     gamma2, beta2, x, out);
}
